// LSTMTrajectoryAutoencoder_11081015624156
// MI455X (gfx1250) — compile-verified, matrix operand order checked by reading, not run on hardware
//
#include <hip/hip_runtime.h>

// ---------------- problem constants ----------------
#define Bx   512
#define Tx   512
#define OBSx 64
#define Hx   256
#define G4x  1024   // 4*H
#define LATx 16
#define MROWS 32    // batch rows per block (2 WMMA row-tiles)

typedef __attribute__((ext_vector_type(16))) __bf16 v16bf;
typedef __attribute__((ext_vector_type(8)))  __bf16 v8bf;
typedef __attribute__((ext_vector_type(8)))  float  v8f;
typedef __attribute__((ext_vector_type(4)))  float  v4f;

__device__ __forceinline__ v8f wmma_bf16(v16bf a, v16bf b, v8f c) {
    return __builtin_amdgcn_wmma_f32_16x16x32_bf16(
        /*neg_a=*/false, a, /*neg_b=*/false, b,
        /*c_mod=*/(short)0, c, /*reuse_a=*/false, /*reuse_b=*/false);
}

__device__ __forceinline__ float sigm(float x) { return 1.0f / (1.0f + __expf(-x)); }
__device__ __forceinline__ float lrelu(float x) { return x > 0.0f ? x : 0.01f * x; }

__device__ __forceinline__ v8f vzero8() {
    v8f z = {0.f,0.f,0.f,0.f,0.f,0.f,0.f,0.f};
    return z;
}

// ---- A-fragment (16x32 bf16) from row-major LDS tile [16][256] at `base` ----
// Per ISA 7.12.2: lane L holds row M=L%16; lanes<16: K in {kt*32+0..7, +16..23},
// lanes>=16: K in {kt*32+8..15, +24..31}.  Two 16B contiguous chunks per lane.
__device__ __forceinline__ v16bf load_a_lds(const __bf16* base, int ktile) {
    const int lane = threadIdx.x & 31;
    const int row  = lane & 15;
    const int kb   = ktile * 32 + ((lane & 16) ? 8 : 0);
    const __bf16* p = base + row * Hx + kb;
    v8bf lo = *(const v8bf*)(p);
    v8bf hi = *(const v8bf*)(p + 16);
    return __builtin_shufflevector(lo, hi, 0,1,2,3,4,5,6,7,8,9,10,11,12,13,14,15);
}

// ---- A-fragment directly from global fp32 x tile (16 batch rows from brow0) ----
__device__ __forceinline__ v16bf load_a_x(const float* __restrict__ x, int brow0, int t, int ktile) {
    const int lane = threadIdx.x & 31;
    const int row  = lane & 15;
    const int kb   = ktile * 32 + ((lane & 16) ? 8 : 0);
    const float* p = x + ((size_t)(brow0 + row) * Tx + t) * OBSx + kb;
    v4f f0 = *(const v4f*)(p);
    v4f f1 = *(const v4f*)(p + 4);
    v4f f2 = *(const v4f*)(p + 16);
    v4f f3 = *(const v4f*)(p + 20);
    v16bf a;
#pragma unroll
    for (int i = 0; i < 4; ++i) {
        a[i]      = (__bf16)f0[i];
        a[i + 4]  = (__bf16)f1[i];
        a[i + 8]  = (__bf16)f2[i];
        a[i + 12] = (__bf16)f3[i];
    }
    return a;
}

// ---- B-fragment: pre-packed per-lane contiguous 32 bytes ----
__device__ __forceinline__ v16bf load_b(const __bf16* __restrict__ packed,
                                        int ntile, int ktile, int ktiles) {
    const int lane = threadIdx.x & 31;
    const __bf16* p = packed + ((((size_t)ntile * ktiles) + ktile) * 32 + lane) * 16;
    return *(const v16bf*)p;
}

// =================== weight pack kernel ===================
// Packs W[K][N] fp32 row-major into bf16 WMMA B-fragment order:
// dst[((nt*ktiles+kt)*32 + lane)*16 + i] = W[k][nt*16 + lane%16],
// k = kt*32 + (lane>=16 ? 8:0) + (i<8 ? i : i+8).
__device__ __forceinline__ void pack_one(const float* __restrict__ W, int K, int N,
                                         __bf16* __restrict__ dst, int idx) {
    const int i    = idx & 15;
    const int L    = (idx >> 4) & 31;
    const int tile = idx >> 9;
    const int ktiles = K >> 5;
    const int kt = tile % ktiles;
    const int nt = tile / ktiles;
    const int n = nt * 16 + (L & 15);
    const int k = kt * 32 + ((L & 16) ? 8 : 0) + (i < 8 ? i : i + 8);
    dst[idx] = (__bf16)W[(size_t)k * N + n];
}

__device__ __forceinline__ void pack_sum(const float* __restrict__ Wa,
                                         const float* __restrict__ Wb, int K, int N,
                                         __bf16* __restrict__ dst, int idx) {
    const int i    = idx & 15;
    const int L    = (idx >> 4) & 31;
    const int tile = idx >> 9;
    const int ktiles = K >> 5;
    const int kt = tile % ktiles;
    const int nt = tile / ktiles;
    const int n = nt * 16 + (L & 15);
    const int k = kt * 32 + ((L & 16) ? 8 : 0) + (i < 8 ? i : i + 8);
    const size_t o = (size_t)k * N + n;
    dst[idx] = (__bf16)(Wa[o] + Wb[o]);
}

__global__ void pack_kernel(const float* __restrict__ Wip,
                            const float* __restrict__ Wi_e, const float* __restrict__ Wh_e,
                            const float* __restrict__ Wi_d, const float* __restrict__ Wh_d,
                            const float* __restrict__ Wop,
                            __bf16* pWip, __bf16* pWi_e, __bf16* pWh_e,
                            __bf16* pWd, __bf16* pWop) {
    const int idx = blockIdx.x * blockDim.x + threadIdx.x;   // 0 .. 262143
    if (idx < 16384) {
        pack_one(Wip, OBSx, Hx, pWip, idx);     // [64,256]
        pack_one(Wop, Hx, OBSx, pWop, idx);     // [256,64]
    }
    pack_one(Wi_e, Hx, G4x, pWi_e, idx);        // [256,1024]
    pack_one(Wh_e, Hx, G4x, pWh_e, idx);
    pack_sum(Wi_d, Wh_d, Hx, G4x, pWd, idx);    // decoder feeds h as x -> sum weights
}

// =================== encoder ===================
// 16 blocks x 512 threads (16 waves). Block owns 32 batch rows (2 row-tiles)
// for all T steps. Wave w owns H-columns [16w,16w+16); its 4 accumulators per
// row-tile are gates i,f,g,o. Each B fragment feeds 2 WMMAs (both row-tiles),
// halving L2 weight traffic vs a 16-row block.
__global__ void __launch_bounds__(512)
enc_kernel(const float* __restrict__ x, const unsigned char* __restrict__ mask,
           const float* __restrict__ bip, const float* __restrict__ b_e,
           const float* __restrict__ Wlp, const float* __restrict__ blp,
           const __bf16* __restrict__ pWip, const __bf16* __restrict__ pWi_e,
           const __bf16* __restrict__ pWh_e,
           float* __restrict__ latent_ws, float* __restrict__ latent_out) {
    __shared__ __bf16 sh_h [MROWS * Hx];   // 16 KB  hidden state (bf16)
    __shared__ __bf16 sh_xp[MROWS * Hx];   // 16 KB  projected input
    __shared__ float  sh_be[G4x];          // 4 KB   gate bias
    __shared__ float  sh_bip[Hx];
    __shared__ unsigned char sh_m[MROWS];

    const int tid  = threadIdx.x;
    const int w    = tid >> 5;          // wave 0..15
    const int lane = tid & 31;
    const int l15  = lane & 15;
    const int hi8  = (lane & 16) ? 8 : 0;
    const int cg   = w * 16 + l15;      // this lane's H column
    const int b0   = blockIdx.x * MROWS;

    for (int i = tid; i < MROWS * Hx; i += 512) sh_h[i] = (__bf16)0.0f;
    for (int i = tid; i < G4x; i += 512)        sh_be[i] = b_e[i];
    if (tid < Hx) sh_bip[tid] = bip[tid];

    float creg[2][8], hreg[2][8];
#pragma unroll
    for (int rt = 0; rt < 2; ++rt)
#pragma unroll
        for (int r = 0; r < 8; ++r) { creg[rt][r] = 0.0f; hreg[rt][r] = 0.0f; }
    __syncthreads();

    const float bi = sh_be[cg], bff = sh_be[256 + cg], bg = sh_be[512 + cg], bo = sh_be[768 + cg];
    const float bxp = sh_bip[cg];

    for (int t = 0; t < Tx; ++t) {
        if (tid < MROWS) sh_m[tid] = mask[(size_t)(b0 + tid) * Tx + t];

        // keep recurrent weights hot (L2) for the upcoming GEMM
        __builtin_prefetch((const void*)(pWh_e + (size_t)w * 8 * 512), 0, 1);

        // ---- xp = leaky_relu(x_t @ Wip + bip); wave w -> xp n-tile w, 2 row-tiles ----
        v8f ax[2] = {vzero8(), vzero8()};
#pragma unroll
        for (int kt = 0; kt < 2; ++kt) {
            v16bf b = load_b(pWip, w, kt, 2);
            ax[0] = wmma_bf16(load_a_x(x, b0,      t, kt), b, ax[0]);
            ax[1] = wmma_bf16(load_a_x(x, b0 + 16, t, kt), b, ax[1]);
        }
#pragma unroll
        for (int rt = 0; rt < 2; ++rt)
#pragma unroll
            for (int r = 0; r < 8; ++r)
                sh_xp[(rt * 16 + r + hi8) * Hx + cg] = (__bf16)lrelu(ax[rt][r] + bxp);
        __syncthreads();

        // ---- z = xp @ Wi_e + h @ Wh_e ; acc[rt][g] = gate g, n-tile g*16+w ----
        v8f acc[2][4];
#pragma unroll
        for (int rt = 0; rt < 2; ++rt)
#pragma unroll
            for (int g = 0; g < 4; ++g) acc[rt][g] = vzero8();

#pragma unroll 2
        for (int kt = 0; kt < 8; ++kt) {
            v16bf a0 = load_a_lds(sh_xp,            kt);
            v16bf a1 = load_a_lds(sh_xp + 16 * Hx,  kt);
#pragma unroll
            for (int g = 0; g < 4; ++g) {
                v16bf b = load_b(pWi_e, g * 16 + w, kt, 8);
                acc[0][g] = wmma_bf16(a0, b, acc[0][g]);
                acc[1][g] = wmma_bf16(a1, b, acc[1][g]);
            }
        }
#pragma unroll 2
        for (int kt = 0; kt < 8; ++kt) {
            v16bf a0 = load_a_lds(sh_h,           kt);
            v16bf a1 = load_a_lds(sh_h + 16 * Hx, kt);
#pragma unroll
            for (int g = 0; g < 4; ++g) {
                v16bf b = load_b(pWh_e, g * 16 + w, kt, 8);
                acc[0][g] = wmma_bf16(a0, b, acc[0][g]);
                acc[1][g] = wmma_bf16(a1, b, acc[1][g]);
            }
        }

        // ---- gates + masked state update (all data thread-private) ----
#pragma unroll
        for (int rt = 0; rt < 2; ++rt)
#pragma unroll
            for (int r = 0; r < 8; ++r) {
                float ii = acc[rt][0][r] + bi;
                float ff = acc[rt][1][r] + bff;
                float gg = acc[rt][2][r] + bg;
                float oo = acc[rt][3][r] + bo;
                float nc = sigm(ff) * creg[rt][r] + sigm(ii) * tanhf(gg);
                float nh = sigm(oo) * tanhf(nc);
                if (sh_m[rt * 16 + r + hi8]) { creg[rt][r] = nc; hreg[rt][r] = nh; }
            }
        __syncthreads();                 // all waves done reading sh_h/sh_xp
#pragma unroll
        for (int rt = 0; rt < 2; ++rt)
#pragma unroll
            for (int r = 0; r < 8; ++r)
                sh_h[(rt * 16 + r + hi8) * Hx + cg] = (__bf16)hreg[rt][r];
        __syncthreads();                 // new h visible to all waves
    }

    // ---- latent = h_f @ Wlp + blp  (32x256 @ 256x16, tiny -> VALU) ----
    {
        const int row = tid >> 4, lc = tid & 15;   // 512 threads = 32 rows x 16 cols
        float s = blp[lc];
        for (int k = 0; k < Hx; ++k)
            s += (float)sh_h[row * Hx + k] * Wlp[k * LATx + lc];
        const size_t o = (size_t)(b0 + row) * LATx + lc;
        latent_ws[o]  = s;
        latent_out[o] = s;
    }
}

// =================== decoder context: ctx = leaky_relu(latent @ Wle + ble) ===================
__global__ void ctx_kernel(const float* __restrict__ latent, const float* __restrict__ Wle,
                           const float* __restrict__ ble, float* __restrict__ ctx) {
    const int idx = blockIdx.x * blockDim.x + threadIdx.x;   // 0 .. 512*256-1
    const int b = idx >> 8, c = idx & 255;
    float s = ble[c];
#pragma unroll
    for (int k = 0; k < LATx; ++k)
        s += latent[b * LATx + k] * Wle[k * Hx + c];
    ctx[idx] = lrelu(s);
}

// =================== decoder ===================
// Same layout as encoder; z = h @ (Wi_d+Wh_d) + b_d, pred = nh @ Wop + bop.
__global__ void __launch_bounds__(512)
dec_kernel(const float* __restrict__ ctx, const float* __restrict__ b_d,
           const float* __restrict__ bop,
           const __bf16* __restrict__ pWd, const __bf16* __restrict__ pWop,
           float* __restrict__ recon) {
    __shared__ __bf16 sh_h[MROWS * Hx];
    __shared__ float  sh_bd[G4x];
    __shared__ float  sh_bop[OBSx];

    const int tid  = threadIdx.x;
    const int w    = tid >> 5;
    const int lane = tid & 31;
    const int l15  = lane & 15;
    const int hi8  = (lane & 16) ? 8 : 0;
    const int cg   = w * 16 + l15;
    const int b0   = blockIdx.x * MROWS;

    for (int i = tid; i < G4x; i += 512) sh_bd[i] = b_d[i];
    if (tid < OBSx) sh_bop[tid] = bop[tid];
    for (int i = tid; i < MROWS * Hx; i += 512)
        sh_h[i] = (__bf16)ctx[(size_t)(b0 + (i >> 8)) * Hx + (i & 255)];

    float creg[2][8], hreg[2][8];
#pragma unroll
    for (int rt = 0; rt < 2; ++rt)
#pragma unroll
        for (int r = 0; r < 8; ++r) {
            const float v = ctx[(size_t)(b0 + rt * 16 + r + hi8) * Hx + cg];
            creg[rt][r] = v; hreg[rt][r] = v;
        }
    __syncthreads();

    const float bi = sh_bd[cg], bff = sh_bd[256 + cg], bg = sh_bd[512 + cg], bo = sh_bd[768 + cg];

    for (int t = 0; t < Tx; ++t) {
        __builtin_prefetch((const void*)(pWd + (size_t)w * 8 * 512), 0, 1);

        v8f acc[2][4];
#pragma unroll
        for (int rt = 0; rt < 2; ++rt)
#pragma unroll
            for (int g = 0; g < 4; ++g) acc[rt][g] = vzero8();

#pragma unroll 2
        for (int kt = 0; kt < 8; ++kt) {
            v16bf a0 = load_a_lds(sh_h,           kt);
            v16bf a1 = load_a_lds(sh_h + 16 * Hx, kt);
#pragma unroll
            for (int g = 0; g < 4; ++g) {
                v16bf b = load_b(pWd, g * 16 + w, kt, 8);
                acc[0][g] = wmma_bf16(a0, b, acc[0][g]);
                acc[1][g] = wmma_bf16(a1, b, acc[1][g]);
            }
        }
#pragma unroll
        for (int rt = 0; rt < 2; ++rt)
#pragma unroll
            for (int r = 0; r < 8; ++r) {
                float ii = acc[rt][0][r] + bi;
                float ff = acc[rt][1][r] + bff;
                float gg = acc[rt][2][r] + bg;
                float oo = acc[rt][3][r] + bo;
                float nc = sigm(ff) * creg[rt][r] + sigm(ii) * tanhf(gg);
                hreg[rt][r] = sigm(oo) * tanhf(nc);
                creg[rt][r] = nc;
            }
        __syncthreads();                 // all waves done reading sh_h
#pragma unroll
        for (int rt = 0; rt < 2; ++rt)
#pragma unroll
            for (int r = 0; r < 8; ++r)
                sh_h[(rt * 16 + r + hi8) * Hx + cg] = (__bf16)hreg[rt][r];
        __syncthreads();                 // new h visible

        // ---- pred = nh @ Wop + bop ; waves 0..3 each own one 16-col tile ----
        if (w < 4) {
            v8f ap[2] = {vzero8(), vzero8()};
#pragma unroll 2
            for (int kt = 0; kt < 8; ++kt) {
                v16bf b = load_b(pWop, w, kt, 8);
                ap[0] = wmma_bf16(load_a_lds(sh_h,           kt), b, ap[0]);
                ap[1] = wmma_bf16(load_a_lds(sh_h + 16 * Hx, kt), b, ap[1]);
            }
            const int col = w * 16 + l15;
            const float bb = sh_bop[col];
#pragma unroll
            for (int rt = 0; rt < 2; ++rt)
#pragma unroll
                for (int r = 0; r < 8; ++r)
                    recon[((size_t)(b0 + rt * 16 + r + hi8) * Tx + t) * OBSx + col] = ap[rt][r] + bb;
        }
    }
}

// =================== host launch ===================
extern "C" void kernel_launch(void* const* d_in, const int* in_sizes, int n_in,
                              void* d_out, int out_size, void* d_ws, size_t ws_size,
                              hipStream_t stream) {
    const float* x    = (const float*)d_in[0];
    const unsigned char* mask = (const unsigned char*)d_in[1];  // jnp bool_ = 1 byte
    const float* Wip  = (const float*)d_in[2];
    const float* bip  = (const float*)d_in[3];
    const float* Wi_e = (const float*)d_in[4];
    const float* Wh_e = (const float*)d_in[5];
    const float* b_e  = (const float*)d_in[6];
    const float* Wlp  = (const float*)d_in[7];
    const float* blp  = (const float*)d_in[8];
    const float* Wle  = (const float*)d_in[9];
    const float* ble  = (const float*)d_in[10];
    const float* Wi_d = (const float*)d_in[11];
    const float* Wh_d = (const float*)d_in[12];
    const float* b_d  = (const float*)d_in[13];
    const float* Wop  = (const float*)d_in[14];
    const float* bop  = (const float*)d_in[15];

    char* ws = (char*)d_ws;
    __bf16* pWip  = (__bf16*)ws; ws += (size_t)16384  * 2;
    __bf16* pWi_e = (__bf16*)ws; ws += (size_t)262144 * 2;
    __bf16* pWh_e = (__bf16*)ws; ws += (size_t)262144 * 2;
    __bf16* pWd   = (__bf16*)ws; ws += (size_t)262144 * 2;
    __bf16* pWop  = (__bf16*)ws; ws += (size_t)16384  * 2;
    float*  ctx   = (float*)ws;  ws += (size_t)Bx * Hx * 4;
    float*  lat   = (float*)ws;  ws += (size_t)Bx * LATx * 4;

    float* recon   = (float*)d_out;                       // [B,T,OBS]
    float* lat_out = recon + (size_t)Bx * Tx * OBSx;      // [B,LAT]

    pack_kernel<<<1024, 256, 0, stream>>>(Wip, Wi_e, Wh_e, Wi_d, Wh_d, Wop,
                                          pWip, pWi_e, pWh_e, pWd, pWop);
    enc_kernel<<<Bx / MROWS, 512, 0, stream>>>(x, mask, bip, b_e, Wlp, blp,
                                               pWip, pWi_e, pWh_e, lat, lat_out);
    ctx_kernel<<<Bx, 256, 0, stream>>>(lat, Wle, ble, ctx);
    dec_kernel<<<Bx / MROWS, 512, 0, stream>>>(ctx, b_d, bop, pWd, pWop, recon);
}
